// RoIHeads_52690658787608
// MI455X (gfx1250) — matrix operand order repaired and checked by reading, not compiled, not run on hardware
//
#include <hip/hip_runtime.h>
#include <hip/hip_bf16.h>
#include <math.h>

// ---------------- problem constants ----------------
#define BATCH   2
#define NPROP   16384
#define DIN     1024
#define HDIM    512
#define NCLS    100
#define KOUT    100
#define NC1     99                       // classes minus background
#define M2      (NPROP * NC1)            // 1,622,016 candidates per frame
#define NSEG    (M2 / 1024)              // 1584 segments per frame
#define LDLOG   128                      // padded logits stride (>=100)
#define LDREG   960                      // padded reg stride (>=900, mult of 64)
#define NEG_INF (-__builtin_inff())
#define BBOX_CLIP 4.135166556742356f     // log(1000/16)

// ---------------- CDNA5 WMMA types ----------------
typedef __attribute__((ext_vector_type(16))) __bf16 v16bf;
typedef __attribute__((ext_vector_type(8)))  __bf16 v8bf;
typedef __attribute__((ext_vector_type(4)))  __bf16 v4bf;
typedef __attribute__((ext_vector_type(8)))  float  v8f;

union BF16Frag { v16bf v; v8bf h[2]; };

// async copy: 16 bytes global -> LDS, tracked by ASYNCcnt
#define ASYNC_B128(ldsOff, gaddr)                                              \
    asm volatile("global_load_async_to_lds_b128 %0, %1, off"                   \
                 :: "v"(ldsOff), "v"(gaddr) : "memory")

// ---------------- fp32 -> bf16 bulk convert (vectorized) ----------------
__global__ __launch_bounds__(256)
void convert_x_kernel(const float* __restrict__ X, __bf16* __restrict__ XB,
                      size_t total) {
    size_t t = ((size_t)blockIdx.x * 256 + threadIdx.x) * 4;
    if (t >= total) return;
    const float4 f = *(const float4*)(X + t);
    v4bf o;
    o[0] = (__bf16)f.x; o[1] = (__bf16)f.y;
    o[2] = (__bf16)f.z; o[3] = (__bf16)f.w;
    *(v4bf*)(XB + t) = o;
}

// ---------------- weight transpose + fp32->bf16 ----------------
// W [K][N] fp32 row-major  ->  WT [Npad][K] bf16 (zero-padded rows)
__global__ __launch_bounds__(256)
void convert_wT_kernel(const float* __restrict__ W, __bf16* __restrict__ WT,
                       int K, int N, int Npad) {
    int t = blockIdx.x * 256 + threadIdx.x;
    int total = Npad * K;
    if (t >= total) return;
    int n = t / K, k = t % K;
    float v = (n < N) ? W[(size_t)k * N + n] : 0.0f;
    WT[t] = (__bf16)v;
}

// ---------------- WMMA GEMM with async double-buffered LDS staging -----
// C[M][ldc] = act( A[M][K] * BT[N][K]^T + bias )
// Block tile 128(M) x 64(N), 8 waves of 32x32, K-step 32.
template <bool OUT_BF16, bool RELU>
__global__ __launch_bounds__(256)
void gemm_wmma_kernel(const __bf16* __restrict__ A,
                      const __bf16* __restrict__ BT,
                      const float* __restrict__ bias, int bias_len,
                      void* __restrict__ Cv,
                      int M, int K, int ldc) {
    __shared__ __bf16 As[2][128][40];   // +8 pad, rows stay 16B aligned
    __shared__ __bf16 Bs[2][64][40];

    const int tid    = threadIdx.x;
    const int wave   = tid >> 5;
    const int lane   = tid & 31;
    const int wm     = wave >> 1;          // 0..3
    const int wn     = wave & 1;           // 0..1
    const int laneN  = lane & 15;
    const int laneHi = lane >> 4;
    const int blockM = blockIdx.y * 128;
    const int blockN = blockIdx.x * 64;

    // staging coordinates: thread loads 8 bf16 (16B) per async op
    const int idx0 = tid * 8;                   // 0..2047
    const int rA0 = idx0 >> 5, cA0 = idx0 & 31; // A rows 0..63
    const int rA1 = rA0 + 64;                   // A rows 64..127
    const int rB  = rA0, cB = cA0;              // B rows 0..63

    const unsigned long long gA0 =
        (unsigned long long)(uintptr_t)(A + (size_t)(blockM + rA0) * K + cA0);
    const unsigned long long gA1 =
        (unsigned long long)(uintptr_t)(A + (size_t)(blockM + rA1) * K + cA0);
    const unsigned long long gB =
        (unsigned long long)(uintptr_t)(BT + (size_t)(blockN + rB) * K + cB);

    const unsigned lA0b0 = (unsigned)(uintptr_t)&As[0][rA0][cA0];
    const unsigned lA0b1 = (unsigned)(uintptr_t)&As[1][rA0][cA0];
    const unsigned lA1b0 = (unsigned)(uintptr_t)&As[0][rA1][cA0];
    const unsigned lA1b1 = (unsigned)(uintptr_t)&As[1][rA1][cA0];
    const unsigned lBb0  = (unsigned)(uintptr_t)&Bs[0][rB][cB];
    const unsigned lBb1  = (unsigned)(uintptr_t)&Bs[1][rB][cB];

    v8f acc[2][2] = {};

    const int nIter = K >> 5;                   // K-steps of 32 (K % 32 == 0)

    // prologue: stage 0 into buffer 0
    ASYNC_B128(lA0b0, gA0);
    ASYNC_B128(lA1b0, gA1);
    ASYNC_B128(lBb0,  gB);

    for (int it = 0; it < nIter; ++it) {
        const int cur = it & 1;
        if (it + 1 < nIter) {
            // issue next stage into the other buffer (64 bytes per K-step)
            const unsigned long long kb = (unsigned long long)(it + 1) * 64ull;
            ASYNC_B128(cur ? lA0b0 : lA0b1, gA0 + kb);
            ASYNC_B128(cur ? lA1b0 : lA1b1, gA1 + kb);
            ASYNC_B128(cur ? lBb0  : lBb1,  gB  + kb);
            // wait until only the 3 just-issued remain -> current stage landed
            asm volatile("s_wait_asynccnt 0x3" ::: "memory");
        } else {
            asm volatile("s_wait_asynccnt 0x0" ::: "memory");
        }
        __syncthreads();

        // ---- fragments per ISA layout ----
        BF16Frag a[2], b[2];
        const int kb  = laneHi * 8;    // A: lanes 0-15 K0-7/16-23, lanes 16-31 K8-15/24-31
        const int kb2 = laneHi * 16;   // (unused) B uses same K order as A
        #pragma unroll
        for (int i = 0; i < 2; ++i) {
            int mr = wm * 32 + i * 16 + laneN;
            a[i].h[0] = *(const v8bf*)&As[cur][mr][kb];
            a[i].h[1] = *(const v8bf*)&As[cur][mr][16 + kb];
        }
        #pragma unroll
        for (int j = 0; j < 2; ++j) {
            int nr = wn * 32 + j * 16 + laneN;
            b[j].h[0] = *(const v8bf*)&Bs[cur][nr][kb];
            b[j].h[1] = *(const v8bf*)&Bs[cur][nr][16 + kb];
        }
        #pragma unroll
        for (int i = 0; i < 2; ++i)
            #pragma unroll
            for (int j = 0; j < 2; ++j)
                acc[i][j] = __builtin_amdgcn_wmma_f32_16x16x32_bf16(
                    false, a[i].v, false, b[j].v, (short)0, acc[i][j], false, false);
        __syncthreads();   // all waves done reading before buffer reuse
    }

    // ---- epilogue: bias + activation + store ----
    #pragma unroll
    for (int i = 0; i < 2; ++i) {
        #pragma unroll
        for (int j = 0; j < 2; ++j) {
            int n = blockN + wn * 32 + j * 16 + laneN;
            float bv = (bias && n < bias_len) ? bias[n] : 0.0f;
            #pragma unroll
            for (int r = 0; r < 8; ++r) {
                int m = blockM + wm * 32 + i * 16 + laneHi * 8 + r;
                float v = acc[i][j][r] + bv;
                if (RELU) v = v > 0.0f ? v : 0.0f;
                if (OUT_BF16)
                    ((__bf16*)Cv)[(size_t)m * ldc + n] = (__bf16)v;
                else
                    ((float*)Cv)[(size_t)m * ldc + n] = v;
            }
        }
    }
}

// ---------------- softmax over 100 classes (in place, stride 128) ----------------
__global__ __launch_bounds__(256)
void softmax_kernel(float* __restrict__ buf, int rows) {
    int r = blockIdx.x * 256 + threadIdx.x;
    if (r >= rows) return;
    float* p = buf + (size_t)r * LDLOG;
    float mx = p[0];
    for (int c = 1; c < NCLS; ++c) mx = fmaxf(mx, p[c]);
    float s = 0.0f;
    for (int c = 0; c < NCLS; ++c) s += __expf(p[c] - mx);
    float inv = 1.0f / s;
    for (int c = 0; c < NCLS; ++c) p[c] = __expf(p[c] - mx) * inv;
}

// ---------------- decode boxes + validity mask (class-major layout) ----------------
__global__ __launch_bounds__(256)
void decode_kernel(const float* __restrict__ proposals,
                   const float* __restrict__ reg,
                   const float* __restrict__ scores,
                   float* __restrict__ xyxy4,
                   float* __restrict__ smask) {
    int t = blockIdx.x * 256 + threadIdx.x;
    if (t >= BATCH * M2) return;
    int b  = t / M2;
    int r2 = t % M2;
    int cp = r2 >> 14;           // class-1 (0..98), class-major
    int n  = r2 & (NPROP - 1);
    int row = b * NPROP + n;

    const float* p = proposals + (size_t)row * 9;
    float w  = p[2] - p[0], h = p[3] - p[1];
    float cx = p[0] + 0.5f * w, cy = p[1] + 0.5f * h;

    const float* rg = reg + (size_t)row * LDREG + (cp + 1) * 9;
    float dx = rg[0] * 0.1f, dy = rg[1] * 0.1f;
    float dw = fminf(rg[2] * 0.2f, BBOX_CLIP);
    float dh = fminf(rg[3] * 0.2f, BBOX_CLIP);
    float pcx = dx * w + cx, pcy = dy * h + cy;
    float pw = __expf(dw) * w, ph = __expf(dh) * h;
    float x1 = pcx - 0.5f * pw, y1 = pcy - 0.5f * ph;
    float x2 = pcx + 0.5f * pw, y2 = pcy + 0.5f * ph;

    float sc = scores[(size_t)row * LDLOG + (cp + 1)];
    bool valid = (sc > 0.05f) && ((x2 - x1) >= 0.01f) && ((y2 - y1) >= 0.01f);

    size_t o = (size_t)b * M2 + r2;
    float4 bx; bx.x = x1; bx.y = y1; bx.z = x2; bx.w = y2;
    *(float4*)(xyxy4 + o * 4) = bx;
    smask[o] = valid ? sc : NEG_INF;
}

// ---------------- per-segment max of smask (1024 entries/segment) ----------------
__global__ __launch_bounds__(128)
void segmax_kernel(const float* __restrict__ smask, float* __restrict__ segmax) {
    int seg = blockIdx.x;                      // 0 .. BATCH*NSEG-1
    int b = seg / NSEG, s = seg % NSEG;
    const float* p = smask + (size_t)b * M2 + (size_t)s * 1024;
    float mx = NEG_INF;
    for (int i = threadIdx.x; i < 1024; i += 128) mx = fmaxf(mx, p[i]);
    __shared__ float red[128];
    red[threadIdx.x] = mx;
    __syncthreads();
    for (int off = 64; off > 0; off >>= 1) {
        if (threadIdx.x < off) red[threadIdx.x] = fmaxf(red[threadIdx.x], red[threadIdx.x + off]);
        __syncthreads();
    }
    if (threadIdx.x == 0) segmax[seg] = red[0];
}

// ---------------- greedy NMS, one frame per workgroup ----------------
__global__ __launch_bounds__(1024)
void nms_kernel(const float* __restrict__ xyxy4,
                float* __restrict__ smask,
                float* __restrict__ segmax,
                const float* __restrict__ scores,
                const float* __restrict__ reg,
                const float* __restrict__ proposals,
                float* __restrict__ out_boxes,
                float* __restrict__ out_scores,
                int* __restrict__ out_labels) {
    const int b = blockIdx.x;
    const int t = threadIdx.x;
    float*       sm  = smask  + (size_t)b * M2;
    const float* bx  = xyxy4  + (size_t)b * M2 * 4;
    float*       sgm = segmax + (size_t)b * NSEG;

    __shared__ float rs[1024];
    __shared__ int   ri[1024];

    for (int k = 0; k < KOUT; ++k) {
        // --- 1) argmax over segment maxima (tie -> smallest segment) ---
        float best = NEG_INF; int bseg = NSEG;
        for (int s = t; s < NSEG; s += 1024) {
            float v = sgm[s];
            if (v > best) { best = v; bseg = s; }
        }
        if (bseg == NSEG) bseg = t;    // all -inf in my slice: keep ordering stable
        rs[t] = best; ri[t] = bseg;
        __syncthreads();
        for (int off = 512; off > 0; off >>= 1) {
            if (t < off) {
                float v2 = rs[t + off]; int i2 = ri[t + off];
                if (v2 > rs[t] || (v2 == rs[t] && i2 < ri[t])) { rs[t] = v2; ri[t] = i2; }
            }
            __syncthreads();
        }
        float vmax = rs[0];
        int   sWin = ri[0];
        __syncthreads();

        // --- 2) exact index inside winning segment (tie -> smallest idx) ---
        int cand = 0x7FFFFFFF;
        {
            int idx = sWin * 1024 + t;
            if (sm[idx] == vmax) cand = idx;
        }
        ri[t] = cand;
        __syncthreads();
        for (int off = 512; off > 0; off >>= 1) {
            if (t < off) ri[t] = min(ri[t], ri[t + off]);
            __syncthreads();
        }
        int chosen = ri[0];
        if (chosen == 0x7FFFFFFF) chosen = 0;   // paranoia; shouldn't happen
        __syncthreads();

        const bool pick = vmax > NEG_INF;
        const int  cp = chosen >> 14;           // class-1
        const int  n  = chosen & (NPROP - 1);

        // --- 3) emit outputs for this iteration ---
        if (t == 0) {
            int row = b * NPROP + n;
            const float4 cb4 = *(const float4*)(bx + (size_t)chosen * 4);
            float* ob = out_boxes + ((size_t)b * KOUT + k) * 9;
            ob[0] = cb4.x; ob[1] = cb4.y; ob[2] = cb4.z; ob[3] = cb4.w;
            const float* p  = proposals + (size_t)row * 9;
            const float* rg = reg + (size_t)row * LDREG + (cp + 1) * 9;
            for (int d = 0; d < 5; ++d) ob[4 + d] = p[4 + d] + rg[4 + d];
            float sraw = scores[(size_t)row * LDLOG + (cp + 1)];
            out_scores[b * KOUT + k] = pick ? sraw : 0.0f;
            out_labels[b * KOUT + k] = pick ? (cp + 1) : -1;
        }

        // --- 4) suppress same-class candidates with IoU > 0.5 ---
        {
            const float4 cb4 = *(const float4*)(bx + (size_t)chosen * 4);
            const float carea = (cb4.z - cb4.x) * (cb4.w - cb4.y);
            const int base = cp << 14;
            for (int j = t; j < NPROP; j += 1024) {
                int idx2 = base + j;
                float s = sm[idx2];
                if (s > NEG_INF) {
                    if (idx2 == chosen) {
                        sm[idx2] = NEG_INF;
                    } else {
                        const float4 c4 = *(const float4*)(bx + (size_t)idx2 * 4);
                        float ix1 = fmaxf(cb4.x, c4.x);
                        float iy1 = fmaxf(cb4.y, c4.y);
                        float ix2 = fminf(cb4.z, c4.z);
                        float iy2 = fminf(cb4.w, c4.w);
                        float inter = fmaxf(ix2 - ix1, 0.0f) * fmaxf(iy2 - iy1, 0.0f);
                        float oarea = (c4.z - c4.x) * (c4.w - c4.y);
                        float iou = inter / (carea + oarea - inter + 1e-9f);
                        if (iou > 0.5f) sm[idx2] = NEG_INF;
                    }
                }
            }
        }
        __syncthreads();

        // --- 5) refresh the 16 segment maxima of the suppressed class ---
        if (t < 512) {
            int g = t >> 5, l = t & 31;
            const float* p = sm + ((size_t)(cp * 16 + g)) * 1024;
            float mx = NEG_INF;
            for (int i = l; i < 1024; i += 32) mx = fmaxf(mx, p[i]);
            rs[t] = mx;
        }
        __syncthreads();
        if (t < 16) {
            float mx = NEG_INF;
            for (int l = 0; l < 32; ++l) mx = fmaxf(mx, rs[t * 32 + l]);
            sgm[cp * 16 + t] = mx;
        }
        __syncthreads();
    }
}

// ---------------- launcher ----------------
extern "C" void kernel_launch(void* const* d_in, const int* in_sizes, int n_in,
                              void* d_out, int out_size, void* d_ws, size_t ws_size,
                              hipStream_t stream) {
    (void)in_sizes; (void)n_in; (void)out_size; (void)ws_size;

    const float* x         = (const float*)d_in[0];   // [B*N, DIN]
    const float* proposals = (const float*)d_in[1];   // [B, N, 9]
    const float* w1 = (const float*)d_in[2];
    const float* b1 = (const float*)d_in[3];
    const float* w2 = (const float*)d_in[4];
    const float* b2 = (const float*)d_in[5];
    const float* wc = (const float*)d_in[6];
    const float* bc = (const float*)d_in[7];
    const float* wr = (const float*)d_in[8];
    const float* br = (const float*)d_in[9];

    const int M = BATCH * NPROP;   // 32768

    // workspace layout (bytes, all 256-aligned)
    char* ws = (char*)d_ws;
    size_t o = 0;
    __bf16* xb  = (__bf16*)(ws + o); o += (size_t)M * DIN * 2;             // 67.1 MB
    __bf16* w1T = (__bf16*)(ws + o); o += (size_t)HDIM * DIN * 2;          //  1.0 MB
    __bf16* w2T = (__bf16*)(ws + o); o += (size_t)HDIM * HDIM * 2;         //  0.5 MB
    __bf16* wcT = (__bf16*)(ws + o); o += (size_t)LDLOG * HDIM * 2;        //  128x512
    __bf16* wrT = (__bf16*)(ws + o); o += (size_t)LDREG * HDIM * 2;        //  960x512
    __bf16* h1  = (__bf16*)(ws + o); o += (size_t)M * HDIM * 2;            // 33.6 MB
    __bf16* h2  = (__bf16*)(ws + o); o += (size_t)M * HDIM * 2;            // 33.6 MB
    float* logits = (float*)(ws + o); o += (size_t)M * LDLOG * 4;          // 16.8 MB (scores after softmax)
    float* regb   = (float*)(ws + o); o += (size_t)M * LDREG * 4;          // 125.8 MB
    float* xyxy4  = (float*)(ws + o); o += (size_t)BATCH * M2 * 4 * 4;     // 51.9 MB
    float* smask  = (float*)(ws + o); o += (size_t)BATCH * M2 * 4;         // 13.0 MB
    float* segmax = (float*)(ws + o); o += (size_t)BATCH * NSEG * 4;

    // 1) input + weight conversion to bf16 (weights transposed to [N][K])
    convert_x_kernel<<<((size_t)M * DIN / 4 + 255) / 256, 256, 0, stream>>>(
        x, xb, (size_t)M * DIN);
    convert_wT_kernel<<<(HDIM * DIN + 255) / 256, 256, 0, stream>>>(w1, w1T, DIN, HDIM, HDIM);
    convert_wT_kernel<<<(HDIM * HDIM + 255) / 256, 256, 0, stream>>>(w2, w2T, HDIM, HDIM, HDIM);
    convert_wT_kernel<<<(LDLOG * HDIM + 255) / 256, 256, 0, stream>>>(wc, wcT, HDIM, NCLS, LDLOG);
    convert_wT_kernel<<<(LDREG * HDIM + 255) / 256, 256, 0, stream>>>(wr, wrT, HDIM, NCLS * 9, LDREG);

    // 2) MLP + heads (WMMA bf16, fp32 accumulate, async LDS staging)
    gemm_wmma_kernel<true,  true ><<<dim3(HDIM / 64, M / 128), 256, 0, stream>>>(
        xb, w1T, b1, HDIM, h1, M, DIN, HDIM);
    gemm_wmma_kernel<true,  true ><<<dim3(HDIM / 64, M / 128), 256, 0, stream>>>(
        h1, w2T, b2, HDIM, h2, M, HDIM, HDIM);
    gemm_wmma_kernel<false, false><<<dim3(LDLOG / 64, M / 128), 256, 0, stream>>>(
        h2, wcT, bc, NCLS, logits, M, HDIM, LDLOG);
    gemm_wmma_kernel<false, false><<<dim3(LDREG / 64, M / 128), 256, 0, stream>>>(
        h2, wrT, br, NCLS * 9, regb, M, HDIM, LDREG);

    // 3) softmax (in place on logits -> scores)
    softmax_kernel<<<(M + 255) / 256, 256, 0, stream>>>(logits, M);

    // 4) decode + validity mask (class-major), then segment maxima
    decode_kernel<<<(BATCH * M2 + 255) / 256, 256, 0, stream>>>(
        proposals, regb, logits, xyxy4, smask);
    segmax_kernel<<<BATCH * NSEG, 128, 0, stream>>>(smask, segmax);

    // 5) greedy NMS + output gather
    float* out_boxes  = (float*)d_out;                         // [B,K,9]
    float* out_scores = out_boxes + (size_t)BATCH * KOUT * 9;  // [B,K]
    int*   out_labels = (int*)(out_scores + (size_t)BATCH * KOUT);
    nms_kernel<<<BATCH, 1024, 0, stream>>>(
        xyxy4, smask, segmax, logits, regb, proposals,
        out_boxes, out_scores, out_labels);
}
